// MultiHeadedAttention_41566693491436
// MI455X (gfx1250) — compile-verified, matrix operand order checked by reading, not run on hardware
//
#include <hip/hip_runtime.h>
#include <hip/hip_bf16.h>

// Problem constants (from reference)
#define SS  2048
#define BB  2
#define DD  1024
#define HH  16
#define DKK 64

typedef __attribute__((ext_vector_type(16))) _Float16 v16h;
typedef __attribute__((ext_vector_type(8)))  float    v8f;

// ---------------------------------------------------------------------------
// WMMA helper: D = A(16x32 f16) * B(32x16 f16) + C(16x16 f32)
// ---------------------------------------------------------------------------
__device__ __forceinline__ v8f wmma(v16h a, v16h b, v8f c) {
    return __builtin_amdgcn_wmma_f32_16x16x32_f16(
        /*neg_a=*/false, a, /*neg_b=*/false, b,
        /*c_mod=*/(short)0, c, /*reuse_a=*/false, /*reuse_b=*/false);
}

// A/B fragment loaders (ISA 16-bit 16x32 layout):
//   lane L: row/col = L&15 ; kbase = (L>>4)*8
//   VGPRs 0..3 hold K = kbase..kbase+7 (16 contiguous bytes)
//   VGPRs 4..7 hold K = 16+kbase..16+kbase+7 (16 contiguous bytes)
// -> each fragment is exactly two 128-bit global loads.
__device__ __forceinline__ v16h frag16(const _Float16* prow) {
    const int kb = ((threadIdx.x >> 4) & 1) * 8;
    union { v16h v; uint4 q[2]; } u;
    u.q[0] = *(const uint4*)(prow + kb);
    u.q[1] = *(const uint4*)(prow + 16 + kb);
    return u.v;
}

// Same fragment, but source data is f32 in memory: four float4 (b128) loads,
// converted to f16 on the fly (v_cvt_pk_f16_f32).
__device__ __forceinline__ v16h frag32(const float* prow) {
    const int kb = ((threadIdx.x >> 4) & 1) * 8;
    float4 x0 = *(const float4*)(prow + kb);
    float4 x1 = *(const float4*)(prow + kb + 4);
    float4 x2 = *(const float4*)(prow + 16 + kb);
    float4 x3 = *(const float4*)(prow + 16 + kb + 4);
    v16h a;
    a[0]  = (_Float16)x0.x; a[1]  = (_Float16)x0.y;
    a[2]  = (_Float16)x0.z; a[3]  = (_Float16)x0.w;
    a[4]  = (_Float16)x1.x; a[5]  = (_Float16)x1.y;
    a[6]  = (_Float16)x1.z; a[7]  = (_Float16)x1.w;
    a[8]  = (_Float16)x2.x; a[9]  = (_Float16)x2.y;
    a[10] = (_Float16)x2.z; a[11] = (_Float16)x2.w;
    a[12] = (_Float16)x3.x; a[13] = (_Float16)x3.y;
    a[14] = (_Float16)x3.z; a[15] = (_Float16)x3.w;
    return a;
}

// ---------------------------------------------------------------------------
// Kernel 1: QKV projection  Y = X(f32, M x D) @ W^T(f32, D x D) + b
//   MODE 0: store f16 head layout   (B,H,S,DK)   (for Q and K)
//   MODE 1: store f16 transposed    (B,H,DK,S)   (for V -> contiguous-K PV B-frags)
// One wave per 16x16 tile; M = S*B = 4096 -> 256x64 tiles -> 2048 blocks x 8 waves.
// ---------------------------------------------------------------------------
template <int MODE>
__global__ __launch_bounds__(256) void proj_qkv(
    const float* __restrict__ X, const float* __restrict__ W,
    const float* __restrict__ bias, _Float16* __restrict__ out)
{
    const int lo   = threadIdx.x & 15;
    const int hi   = (threadIdx.x >> 4) & 1;
    const int tile = blockIdx.x * 8 + (threadIdx.x >> 5);
    const int tn = tile & 63;          // N/16 = 64
    const int tm = tile >> 6;          // M/16 = 256
    const int m0 = tm * 16, n0 = tn * 16;

    const float* xrow = X + (size_t)(m0 + lo) * DD;   // A: lane = output row
    const float* wrow = W + (size_t)(n0 + lo) * DD;   // B = W^T: lane = output col -> W row
    v8f acc = {};
    for (int k0 = 0; k0 < DD; k0 += 32)
        acc = wmma(frag32(xrow + k0), frag32(wrow + k0), acc);

    const float bv  = bias[n0 + lo];
    const int   col = n0 + lo;
    const int   h   = col >> 6;        // / DK
    const int   dk  = col & 63;
#pragma unroll
    for (int r = 0; r < 8; ++r) {
        const int rg = m0 + r + 8 * hi;          // row in (s*B + b)
        const int s  = rg >> 1;                  // B = 2
        const int b  = rg & 1;
        const _Float16 y = (_Float16)(acc[r] + bv);
        if (MODE == 0)
            out[(((size_t)(b * HH + h)) * SS + s) * DKK + dk] = y;   // (B,H,S,DK)
        else
            out[(((size_t)(b * HH + h)) * DKK + dk) * SS + s] = y;   // (B,H,DK,S)
    }
}

// ---------------------------------------------------------------------------
// Kernel 2: flash attention per (b,h).  Block = 8 waves, each wave owns a
// 16-row query tile; online softmax over 32 keys/iteration:
//   4 score WMMAs (2 key-frags x 2 DK-chunks) + 4 PV WMMAs.
// Saves per-row (max, sum) stats for the attention-mean recompute kernel.
// ---------------------------------------------------------------------------
__global__ __launch_bounds__(256) void attn_flash(
    const _Float16* __restrict__ qh, const _Float16* __restrict__ kh,
    const _Float16* __restrict__ vt, const float* __restrict__ mask,
    _Float16* __restrict__ xatt, float* __restrict__ stats)
{
    __shared__ _Float16 pbuf[8][16][40];   // per-wave P staging (C-layout -> A-layout)

    const int wave = threadIdx.x >> 5;
    const int lo   = threadIdx.x & 15;
    const int hi   = (threadIdx.x >> 4) & 1;
    const int bh   = blockIdx.x >> 4;      // grid = (B*H) * (S/128)
    const int qblk = blockIdx.x & 15;
    const int b    = bh >> 4;              // / H
    const int h    = bh & 15;
    const int q0   = qblk * 128 + wave * 16;

    // Q A-fragments (persistent): DK=64 -> two 32-wide K chunks
    const _Float16* qrow = qh + ((size_t)bh * SS + q0 + lo) * DKK;
    const v16h qa0 = frag16(qrow);
    const v16h qa1 = frag16(qrow + 32);

    float mrow[8], lrow[8];
    v8f   o[4] = {};
#pragma unroll
    for (int r = 0; r < 8; ++r) { mrow[r] = -1e30f; lrow[r] = 0.f; }

    const float  scale = 0.125f;                         // 1/sqrt(DK)
    const float* mbase = mask + ((size_t)b * SS + q0) * SS;

    for (int j0 = 0; j0 < SS; j0 += 32) {
        // ---- scores for keys [j0, j0+32): two 16x16 C-frags ----
        v8f sc[2];
#pragma unroll
        for (int f = 0; f < 2; ++f) {
            const _Float16* krow = kh + ((size_t)bh * SS + j0 + f * 16 + lo) * DKK;
            v8f c = {};
            c = wmma(qa0, frag16(krow), c);
            c = wmma(qa1, frag16(krow + 32), c);
            sc[f] = c;
        }
        // ---- scale + mask ----
        float s0[8], s1[8];
#pragma unroll
        for (int r = 0; r < 8; ++r) {
            const int m = r + 8 * hi;
            s0[r] = sc[0][r] * scale + mbase[(size_t)m * SS + j0 + lo];
            s1[r] = sc[1][r] * scale + mbase[(size_t)m * SS + j0 + 16 + lo];
        }
        // ---- online softmax: row max across both frags + 16 lanes ----
        float mnew[8];
#pragma unroll
        for (int r = 0; r < 8; ++r) mnew[r] = fmaxf(mrow[r], fmaxf(s0[r], s1[r]));
#pragma unroll
        for (int d = 1; d < 16; d <<= 1)
#pragma unroll
            for (int r = 0; r < 8; ++r) mnew[r] = fmaxf(mnew[r], __shfl_xor(mnew[r], d, 32));

        float p0[8], p1[8], corr[8], psum[8];
#pragma unroll
        for (int r = 0; r < 8; ++r) {
            corr[r] = __expf(mrow[r] - mnew[r]);
            p0[r]   = __expf(s0[r] - mnew[r]);
            p1[r]   = __expf(s1[r] - mnew[r]);
            psum[r] = p0[r] + p1[r];
            mrow[r] = mnew[r];
        }
#pragma unroll
        for (int d = 1; d < 16; d <<= 1)
#pragma unroll
            for (int r = 0; r < 8; ++r) psum[r] += __shfl_xor(psum[r], d, 32);
#pragma unroll
        for (int r = 0; r < 8; ++r) lrow[r] = lrow[r] * corr[r] + psum[r];
        // ---- rescale O ----
#pragma unroll
        for (int c = 0; c < 4; ++c)
#pragma unroll
            for (int r = 0; r < 8; ++r) o[c][r] *= corr[r];
        // ---- P: C-layout -> A-layout via per-wave LDS (in-order within wave) ----
#pragma unroll
        for (int r = 0; r < 8; ++r) {
            pbuf[wave][r + 8 * hi][lo]      = (_Float16)p0[r];
            pbuf[wave][r + 8 * hi][16 + lo] = (_Float16)p1[r];
        }
        v16h pa;
        {
            const int kb = hi * 8;
            const _Float16* prow = &pbuf[wave][lo][0];
            union { v16h v; uint2 d[4]; } u;       // 80B row offset is 8B-aligned
            u.d[0] = *(const uint2*)(prow + kb);
            u.d[1] = *(const uint2*)(prow + kb + 4);
            u.d[2] = *(const uint2*)(prow + 16 + kb);
            u.d[3] = *(const uint2*)(prow + 16 + kb + 4);
            pa = u.v;
        }
        // ---- O += P @ V ; V^T layout makes B-frag K contiguous ----
#pragma unroll
        for (int c = 0; c < 4; ++c) {
            const _Float16* vrow = vt + ((size_t)bh * DKK + c * 16 + lo) * SS + j0;
            o[c] = wmma(pa, frag16(vrow), o[c]);
        }
    }

    // ---- normalize, store x_att (S,B,D) f16 and per-row softmax stats ----
    float inv[8];
#pragma unroll
    for (int r = 0; r < 8; ++r) inv[r] = 1.0f / lrow[r];
#pragma unroll
    for (int c = 0; c < 4; ++c)
#pragma unroll
        for (int r = 0; r < 8; ++r) {
            const int s   = q0 + r + 8 * hi;
            const int col = h * DKK + c * 16 + lo;
            xatt[((size_t)s * BB + b) * DD + col] = (_Float16)(o[c][r] * inv[r]);
        }
    if (lo == 0) {
#pragma unroll
        for (int r = 0; r < 8; ++r) {
            const int s = q0 + r + 8 * hi;
            stats[((size_t)bh * SS + s) * 2 + 0] = mrow[r];
            stats[((size_t)bh * SS + s) * 2 + 1] = lrow[r];
        }
    }
}

// ---------------------------------------------------------------------------
// Kernel 3: output projection  out = x_att(f16) @ Wo^T + bo   (f32 out, S,B,D)
// ---------------------------------------------------------------------------
__global__ __launch_bounds__(256) void proj_out(
    const _Float16* __restrict__ X, const float* __restrict__ W,
    const float* __restrict__ bias, float* __restrict__ out)
{
    const int lo   = threadIdx.x & 15;
    const int hi   = (threadIdx.x >> 4) & 1;
    const int tile = blockIdx.x * 8 + (threadIdx.x >> 5);
    const int tn = tile & 63;
    const int tm = tile >> 6;
    const int m0 = tm * 16, n0 = tn * 16;

    const _Float16* xrow = X + (size_t)(m0 + lo) * DD;
    const float*    wrow = W + (size_t)(n0 + lo) * DD;
    v8f acc = {};
    for (int k0 = 0; k0 < DD; k0 += 32)
        acc = wmma(frag16(xrow + k0), frag32(wrow + k0), acc);

    const float bv = bias[n0 + lo];
#pragma unroll
    for (int r = 0; r < 8; ++r) {
        const int rg = m0 + r + 8 * hi;
        out[(size_t)rg * DD + n0 + lo] = acc[r] + bv;
    }
}

// ---------------------------------------------------------------------------
// Kernel 4: attn = mean over heads of softmax(QK^T/sqrt+mask).
// Recompute scores with WMMA (Q/K tiles are L2-resident), normalize with the
// stored (max,sum) stats. Grid = B * (S/16) * (S/128), 8 waves/block.
// ---------------------------------------------------------------------------
__global__ __launch_bounds__(256) void attn_mean(
    const _Float16* __restrict__ qh, const _Float16* __restrict__ kh,
    const float* __restrict__ mask, const float* __restrict__ stats,
    float* __restrict__ attn)
{
    const int wave = threadIdx.x >> 5;
    const int lo   = threadIdx.x & 15;
    const int hi   = (threadIdx.x >> 4) & 1;
    int idx = blockIdx.x;
    const int tblk = idx & 15;  idx >>= 4;   // S/128
    const int sblk = idx & 127; idx >>= 7;   // S/16
    const int b    = idx;
    const int s0   = sblk * 16;
    const int t0   = tblk * 128 + wave * 16;

    float maskv[8], acc[8];
#pragma unroll
    for (int r = 0; r < 8; ++r) {
        maskv[r] = mask[((size_t)b * SS + s0 + r + 8 * hi) * SS + t0 + lo];
        acc[r]   = 0.f;
    }
    const float scale = 0.125f;

    for (int h = 0; h < HH; ++h) {
        const int bh = b * HH + h;
        const _Float16* qrow = qh + ((size_t)bh * SS + s0 + lo) * DKK;
        const _Float16* krow = kh + ((size_t)bh * SS + t0 + lo) * DKK;
        v8f c = {};
        c = wmma(frag16(qrow),      frag16(krow),      c);
        c = wmma(frag16(qrow + 32), frag16(krow + 32), c);
#pragma unroll
        for (int r = 0; r < 8; ++r) {
            const float* st = stats + ((size_t)bh * SS + s0 + r + 8 * hi) * 2;
            acc[r] += __expf(c[r] * scale + maskv[r] - st[0]) / st[1];
        }
    }
    const float invH = 1.0f / HH;
#pragma unroll
    for (int r = 0; r < 8; ++r)
        attn[((size_t)b * SS + s0 + r + 8 * hi) * SS + t0 + lo] = acc[r] * invH;
}

// ---------------------------------------------------------------------------
// Launcher
// ---------------------------------------------------------------------------
extern "C" void kernel_launch(void* const* d_in, const int* in_sizes, int n_in,
                              void* d_out, int out_size, void* d_ws, size_t ws_size,
                              hipStream_t stream) {
    const float* query = (const float*)d_in[0];
    const float* key   = (const float*)d_in[1];
    const float* value = (const float*)d_in[2];
    const float* mask  = (const float*)d_in[3];
    const float* Wq = (const float*)d_in[4];  const float* bq = (const float*)d_in[5];
    const float* Wk = (const float*)d_in[6];  const float* bk = (const float*)d_in[7];
    const float* Wv = (const float*)d_in[8];  const float* bv = (const float*)d_in[9];
    const float* Wo = (const float*)d_in[10]; const float* bo = (const float*)d_in[11];

    float* out  = (float*)d_out;                       // (S,B,D)
    float* attn = out + (size_t)SS * BB * DD;          // (B,S,S)

    // Workspace layout (~34.1 MB total)
    const size_t headElems = (size_t)BB * HH * SS * DKK;   // 4,194,304
    _Float16* qh   = (_Float16*)d_ws;                  // (B,H,S,DK)  f16
    _Float16* kh   = qh + headElems;                   // (B,H,S,DK)  f16
    _Float16* vt   = kh + headElems;                   // (B,H,DK,S)  f16
    _Float16* xatt = vt + headElems;                   // (S,B,D)     f16
    float*    stats = (float*)(xatt + (size_t)SS * BB * DD); // (B,H,S,2) f32

    proj_qkv<0><<<2048, 256, 0, stream>>>(query, Wq, bq, qh);
    proj_qkv<0><<<2048, 256, 0, stream>>>(key,   Wk, bk, kh);
    proj_qkv<1><<<2048, 256, 0, stream>>>(value, Wv, bv, vt);
    attn_flash<<<512, 256, 0, stream>>>(qh, kh, vt, mask, xatt, stats);
    proj_out<<<2048, 256, 0, stream>>>(xatt, Wo, bo, out);
    attn_mean<<<4096, 256, 0, stream>>>(qh, kh, mask, stats, attn);
}
